// MixtralBlock_16733192585652
// MI455X (gfx1250) — matrix operand order repaired and checked by reading, not compiled, not run on hardware
//
#include <hip/hip_runtime.h>
#include <hip/hip_bf16.h>

// ---------------------------------------------------------------------------
// CDNA5 / gfx1250 types for WMMA
// ---------------------------------------------------------------------------
typedef __bf16 v16bf __attribute__((ext_vector_type(16)));
typedef float  v8f   __attribute__((ext_vector_type(8)));
typedef int    v4i_g __attribute__((vector_size(16)));   // matches builtin param

#if defined(__has_builtin)
  #if __has_builtin(__builtin_amdgcn_global_load_async_to_lds_b128)
    #define USE_ASYNC_LDS 1
  #endif
  #if __has_builtin(__builtin_amdgcn_s_wait_asynccnt)
    #define HAVE_WAIT_ASYNC 1
  #endif
#endif

__device__ __forceinline__ v8f wmma_bf16(v16bf a, v16bf b, v8f c) {
  // v_wmma_f32_16x16x32_bf16  D = A*B + C   (f32 accumulate)
  return __builtin_amdgcn_wmma_f32_16x16x32_bf16(
      /*neg_a=*/false, a, /*neg_b=*/false, b,
      /*c_mod=*/(short)0, c, /*reuse_a=*/false, /*reuse_b=*/false);
}

// 16-byte global -> LDS copy; async DMA (ASYNCcnt) when available.
__device__ __forceinline__ void async_copy16(const void* gp, void* lp) {
#ifdef USE_ASYNC_LDS
  __builtin_amdgcn_global_load_async_to_lds_b128(
      (__attribute__((address_space(1))) v4i_g*)gp,
      (__attribute__((address_space(3))) v4i_g*)lp, 0, 0);
#else
  *(uint4*)lp = *(const uint4*)gp;
#endif
}
__device__ __forceinline__ void async_wait_all() {
#ifdef USE_ASYNC_LDS
  #ifdef HAVE_WAIT_ASYNC
    __builtin_amdgcn_s_wait_asynccnt(0);
  #else
    asm volatile("s_wait_asynccnt 0" ::: "memory");
  #endif
#endif
}

union FragCvt { uint4 u[2]; v16bf v; };

// A-operand fragment (16x32 bf16). Layout: lane L (lo=L%16 is row M),
// hi=L/16: elements 0..15 hold K = 16*hi .. 16*hi+15 (matches B).
// `base` = LDS row start at this k-step.
__device__ __forceinline__ v16bf ldsA_frag(const __bf16* base, int hi) {
  FragCvt c;
  c.u[0] = *(const uint4*)(base + hi * 16);
  c.u[1] = *(const uint4*)(base + hi * 16 + 8);
  return c.v;
}

// B-operand fragment (32x16 bf16), B stored transposed in LDS as [N][K]:
// lane L: column N = L%16, elements 0..15 hold K = 16*(L/16) .. +15
__device__ __forceinline__ v16bf ldsB_frag(const __bf16* p) {
  FragCvt c;
  c.u[0] = ((const uint4*)p)[0];
  c.u[1] = ((const uint4*)p)[1];
  return c.v;
}

__device__ __forceinline__ float hmax16(float v) {
  #pragma unroll
  for (int d = 1; d < 16; d <<= 1) v = fmaxf(v, __shfl_xor(v, d, 32));
  return v;
}
__device__ __forceinline__ float hsum16(float v) {
  #pragma unroll
  for (int d = 1; d < 16; d <<= 1) v += __shfl_xor(v, d, 32);
  return v;
}

// ---------------------------------------------------------------------------
// Problem constants (match reference)
// ---------------------------------------------------------------------------
#define BB   2
#define TT   2048
#define CC   1024
#define HH   16
#define KVH  4
#define DD   64
#define FF   3584
#define EE   8
#define WINW 1024
#define MROWS (BB*TT)   /* 4096 */

// ---------------------------------------------------------------------------
// RMSNorm (one block per row); optionally copies raw input row to copy_out
// ---------------------------------------------------------------------------
__global__ __launch_bounds__(256) void k_rmsnorm(
    const float* __restrict__ x, const float* __restrict__ w,
    float* __restrict__ y, float* __restrict__ copy_out, int Cdim) {
  int row = blockIdx.x, tid = threadIdx.x;
  __shared__ float red[256];
  const float* xr = x + (size_t)row * Cdim;
  float ss = 0.f;
  for (int c = tid; c < Cdim; c += 256) { float v = xr[c]; ss += v * v; }
  red[tid] = ss; __syncthreads();
  for (int s = 128; s > 0; s >>= 1) { if (tid < s) red[tid] += red[tid + s]; __syncthreads(); }
  float inv = rsqrtf(red[0] / (float)Cdim + 1e-5f);
  for (int c = tid; c < Cdim; c += 256) {
    float v = xr[c];
    y[(size_t)row * Cdim + c] = v * inv * w[c];
    if (copy_out) copy_out[(size_t)row * Cdim + c] = v;
  }
}

// ---------------------------------------------------------------------------
// RoPE: fp32 in -> bf16 out, layout [rows][Nh*64], cos/sin [T][32]
// ---------------------------------------------------------------------------
__global__ __launch_bounds__(256) void k_rope(
    const float* __restrict__ in, __bf16* __restrict__ out,
    const float* __restrict__ cosT, const float* __restrict__ sinT,
    int rows, int Nh) {
  int g = blockIdx.x * blockDim.x + threadIdx.x;
  int total = rows * Nh * 32;
  if (g >= total) return;
  int i = g & 31;
  int h = (g >> 5) % Nh;
  int row = g / (32 * Nh);
  int t = row % TT;
  int stride = Nh * 64;
  size_t base = (size_t)row * stride + h * 64 + 2 * i;
  float x1 = in[base], x2 = in[base + 1];
  float c = cosT[t * 32 + i], s = sinT[t * 32 + i];
  out[base]     = (__bf16)(x1 * c - x2 * s);
  out[base + 1] = (__bf16)(x1 * s + x2 * c);
}

// ---------------------------------------------------------------------------
// Tiled WMMA GEMM: C[M,N] = A[M,K] @ B[K,N]   (fp32 in, bf16 WMMA, fp32 acc)
// 128 threads = 4 waves; block tile 64x64, K-step 32; wave -> 32x32 quadrant.
// Register-level double buffering: next K-tile global loads issued before the
// WMMA pass on the current tile so HBM/L2 latency overlaps matrix compute.
// MODE: 0 store, 1 store+resid, 2 silu-gate (B=W1, B2=W2, C=g), 3 C+=scale[m]*acc
// ---------------------------------------------------------------------------
enum { GM_STORE = 0, GM_ADD = 1, GM_GATE = 2, GM_SCALE_ADD = 3 };

template <int MODE>
__global__ __launch_bounds__(128) void k_gemm(
    const float* __restrict__ A, const float* __restrict__ B,
    const float* __restrict__ B2, float* __restrict__ C,
    const float* __restrict__ extra, int M, int N, int Kt,
    int scale_stride, int scale_off) {
  __shared__ alignas(16) __bf16 sA[64 * 32];
  __shared__ alignas(16) __bf16 sB[64 * 32];
  __shared__ alignas(16) __bf16 sB2[(MODE == GM_GATE) ? 64 * 32 : 8];

  const int tid = threadIdx.x;
  const int wave = tid >> 5, lane = tid & 31;
  const int lo = lane & 15, hi = lane >> 4;
  const int m0 = blockIdx.y * 64, n0 = blockIdx.x * 64;
  const int qm = (wave >> 1) * 32, qn = (wave & 1) * 32;

  auto storeA = [&](int it, float4 f) {
    int i = tid + it * 128;
    int r = i >> 3, cg = (i & 7) * 4;
    union { __bf16 h[4]; uint2 u; } pk;
    pk.h[0] = (__bf16)f.x; pk.h[1] = (__bf16)f.y;
    pk.h[2] = (__bf16)f.z; pk.h[3] = (__bf16)f.w;
    *(uint2*)(sA + r * 32 + cg) = pk.u;
  };
  auto storeB = [&](__bf16* dst, int it, float4 f) {
    int i = tid + it * 128;
    int kr = i >> 4, ng = (i & 15) * 4;
    dst[(ng + 0) * 32 + kr] = (__bf16)f.x;
    dst[(ng + 1) * 32 + kr] = (__bf16)f.y;
    dst[(ng + 2) * 32 + kr] = (__bf16)f.z;
    dst[(ng + 3) * 32 + kr] = (__bf16)f.w;
  };
  auto loadA = [&](int it, int k0) -> float4 {
    int i = tid + it * 128;
    int r = i >> 3, cg = (i & 7) * 4;
    return *(const float4*)(A + (size_t)(m0 + r) * Kt + k0 + cg);
  };
  auto loadB = [&](const float* src, int it, int k0) -> float4 {
    int i = tid + it * 128;
    int kr = i >> 4, ng = (i & 15) * 4;
    return *(const float4*)(src + (size_t)(k0 + kr) * N + n0 + ng);
  };

  v8f acc[2][2] = {};
  v8f acc2[2][2] = {};

  // prologue: stage K-tile 0
  #pragma unroll
  for (int it = 0; it < 4; ++it) storeA(it, loadA(it, 0));
  #pragma unroll
  for (int it = 0; it < 4; ++it) storeB(sB, it, loadB(B, it, 0));
  if constexpr (MODE == GM_GATE) {
    #pragma unroll
    for (int it = 0; it < 4; ++it) storeB(sB2, it, loadB(B2, it, 0));
  }
  __syncthreads();

  float4 ra[4], rb[4], rb2[4];
  for (int k0 = 0; k0 < Kt; k0 += 32) {
    const bool nxt = (k0 + 32 < Kt);
    if (nxt) {
      #pragma unroll
      for (int it = 0; it < 4; ++it) ra[it] = loadA(it, k0 + 32);
      #pragma unroll
      for (int it = 0; it < 4; ++it) rb[it] = loadB(B, it, k0 + 32);
      if constexpr (MODE == GM_GATE) {
        #pragma unroll
        for (int it = 0; it < 4; ++it) rb2[it] = loadB(B2, it, k0 + 32);
      }
    }

    // WMMA pass on current LDS tile
    v16bf af0 = ldsA_frag(sA + (qm + lo) * 32, hi);
    v16bf af1 = ldsA_frag(sA + (qm + 16 + lo) * 32, hi);
    #pragma unroll
    for (int ni = 0; ni < 2; ++ni) {
      v16bf bf = ldsB_frag(sB + (qn + ni * 16 + lo) * 32 + hi * 16);
      acc[0][ni] = wmma_bf16(af0, bf, acc[0][ni]);
      acc[1][ni] = wmma_bf16(af1, bf, acc[1][ni]);
      if constexpr (MODE == GM_GATE) {
        v16bf b2 = ldsB_frag(sB2 + (qn + ni * 16 + lo) * 32 + hi * 16);
        acc2[0][ni] = wmma_bf16(af0, b2, acc2[0][ni]);
        acc2[1][ni] = wmma_bf16(af1, b2, acc2[1][ni]);
      }
    }
    __syncthreads();

    if (nxt) {
      #pragma unroll
      for (int it = 0; it < 4; ++it) storeA(it, ra[it]);
      #pragma unroll
      for (int it = 0; it < 4; ++it) storeB(sB, it, rb[it]);
      if constexpr (MODE == GM_GATE) {
        #pragma unroll
        for (int it = 0; it < 4; ++it) storeB(sB2, it, rb2[it]);
      }
      __syncthreads();
    }
  }

  // --- epilogue: C/D layout: element v -> row m = v + 8*hi, col n = lo ---
  #pragma unroll
  for (int mi = 0; mi < 2; ++mi)
    #pragma unroll
    for (int ni = 0; ni < 2; ++ni)
      #pragma unroll
      for (int v = 0; v < 8; ++v) {
        int gm = m0 + qm + mi * 16 + v + 8 * hi;
        int gn = n0 + qn + ni * 16 + lo;
        size_t idx = (size_t)gm * N + gn;
        float a = acc[mi][ni][v];
        if constexpr (MODE == GM_STORE) {
          C[idx] = a;
        } else if constexpr (MODE == GM_ADD) {
          C[idx] = a + extra[idx];
        } else if constexpr (MODE == GM_GATE) {
          float h2 = acc2[mi][ni][v];
          float sg = h2 / (1.f + __expf(-h2));   // silu(x@W2)
          C[idx] = sg * a;                        // * (x@W1)
        } else {  // GM_SCALE_ADD
          C[idx] += extra[(size_t)gm * scale_stride + scale_off] * a;
        }
      }
}

// ---------------------------------------------------------------------------
// Windowed causal flash attention with GQA.
// Block = (qtile of 64, head, batch); 128 threads = 4 waves.
// Wave w owns query strip rows [w*16, w*16+16) over the full 64-key block,
// so online-softmax row stats stay wave-local.
// Q/K tiles staged via GLOBAL_LOAD_ASYNC_TO_LDS (ASYNCcnt) when available.
// ---------------------------------------------------------------------------
__global__ __launch_bounds__(128) void k_attn(
    const __bf16* __restrict__ Qb,   // [B*T][H*64]   bf16 (post-RoPE)
    const __bf16* __restrict__ Kb,   // [B*T][KV*64]  bf16 (post-RoPE)
    const float*  __restrict__ Vf,   // [B*T][KV*64]  f32
    float* __restrict__ O) {         // [B*T][H*64]   f32
  __shared__ alignas(16) __bf16 sQ[64 * 64];
  __shared__ alignas(16) __bf16 sK[64 * 64];
  __shared__ alignas(16) __bf16 sVt[64 * 64];  // [d][key]
  __shared__ alignas(16) __bf16 sP[64 * 64];   // per-wave 16-row strips

  const int tid = threadIdx.x;
  const int wave = tid >> 5, lane = tid & 31;
  const int lo = lane & 15, hi = lane >> 4;
  const int qt = blockIdx.x, h = blockIdx.y, b = blockIdx.z;
  const int kvh = h / (HH / KVH);
  const int brow = b * TT;
  const float scale = 0.125f;  // 1/sqrt(64)

  // stage Q tile (64x64 bf16, async DMA to LDS)
  #pragma unroll
  for (int it = 0; it < 4; ++it) {
    int i = tid + it * 128;
    int r = i >> 3, cg = i & 7;
    async_copy16(Qb + (size_t)(brow + qt * 64 + r) * (HH * 64) + h * 64 + cg * 8,
                 sQ + r * 64 + cg * 8);
  }

  float m_st[8], l_st[8];
  #pragma unroll
  for (int v = 0; v < 8; ++v) { m_st[v] = -1e30f; l_st[v] = 0.f; }
  v8f acc_o[4] = {};

  const int kb_lo = (qt > 16) ? (qt - 16) : 0;
  for (int kb = kb_lo; kb <= qt; ++kb) {
    // stage K block [key][d] (async DMA to LDS)
    #pragma unroll
    for (int it = 0; it < 4; ++it) {
      int i = tid + it * 128;
      int r = i >> 3, cg = i & 7;
      async_copy16(Kb + (size_t)(brow + kb * 64 + r) * (KVH * 64) + kvh * 64 + cg * 8,
                   sK + r * 64 + cg * 8);
    }
    // stage V block transposed -> sVt[d][key], f32 -> bf16 (needs VALU convert)
    #pragma unroll
    for (int it = 0; it < 8; ++it) {
      int i = tid + it * 128;
      int r = i >> 4, ng = (i & 15) * 4;
      float4 f = *(const float4*)(Vf + (size_t)(brow + kb * 64 + r) * (KVH * 64) + kvh * 64 + ng);
      sVt[(ng + 0) * 64 + r] = (__bf16)f.x;
      sVt[(ng + 1) * 64 + r] = (__bf16)f.y;
      sVt[(ng + 2) * 64 + r] = (__bf16)f.z;
      sVt[(ng + 3) * 64 + r] = (__bf16)f.w;
    }
    async_wait_all();
    __syncthreads();

    // S strip = Q_strip(16x64) @ K^T : K in LDS as [key][d] == B^T layout
    v8f acc_s[4] = {};
    #pragma unroll
    for (int s = 0; s < 2; ++s) {
      v16bf a = ldsA_frag(sQ + (wave * 16 + lo) * 64 + s * 32, hi);
      #pragma unroll
      for (int ni = 0; ni < 4; ++ni) {
        v16bf bf = ldsB_frag(sK + (ni * 16 + lo) * 64 + s * 32 + hi * 16);
        acc_s[ni] = wmma_bf16(a, bf, acc_s[ni]);
      }
    }

    // mask + scale; row max
    float mx[8];
    #pragma unroll
    for (int v = 0; v < 8; ++v) mx[v] = -1e30f;
    #pragma unroll
    for (int ni = 0; ni < 4; ++ni) {
      int j = kb * 64 + ni * 16 + lo;
      #pragma unroll
      for (int v = 0; v < 8; ++v) {
        int iq = qt * 64 + wave * 16 + v + 8 * hi;
        bool ok = (j <= iq) && (j >= iq - (WINW - 1));
        float sv = ok ? acc_s[ni][v] * scale : -1e30f;
        acc_s[ni][v] = sv;
        mx[v] = fmaxf(mx[v], sv);
      }
    }
    float fs[8], mnew[8];
    #pragma unroll
    for (int v = 0; v < 8; ++v) {
      mx[v] = hmax16(mx[v]);
      mnew[v] = fmaxf(m_st[v], mx[v]);
      fs[v] = __expf(m_st[v] - mnew[v]);
    }
    // probabilities (explicit mask to avoid exp(0) on fully-masked rows)
    #pragma unroll
    for (int ni = 0; ni < 4; ++ni) {
      int j = kb * 64 + ni * 16 + lo;
      #pragma unroll
      for (int v = 0; v < 8; ++v) {
        int iq = qt * 64 + wave * 16 + v + 8 * hi;
        bool ok = (j <= iq) && (j >= iq - (WINW - 1));
        acc_s[ni][v] = ok ? __expf(acc_s[ni][v] - mnew[v]) : 0.f;
      }
    }
    #pragma unroll
    for (int v = 0; v < 8; ++v) {
      float r = acc_s[0][v] + acc_s[1][v] + acc_s[2][v] + acc_s[3][v];
      l_st[v] = l_st[v] * fs[v] + hsum16(r);
      m_st[v] = mnew[v];
    }
    // rescale O
    #pragma unroll
    for (int ni = 0; ni < 4; ++ni)
      #pragma unroll
      for (int v = 0; v < 8; ++v) acc_o[ni][v] *= fs[v];

    // write P strip to LDS (accum layout -> A-operand layout), wave-local
    #pragma unroll
    for (int ni = 0; ni < 4; ++ni)
      #pragma unroll
      for (int v = 0; v < 8; ++v)
        sP[(wave * 16 + v + 8 * hi) * 64 + ni * 16 + lo] = (__bf16)acc_s[ni][v];
    asm volatile("s_wait_dscnt 0" ::: "memory");

    // O += P(16x64) @ V(64x64) : V^T in LDS as [d][key] == B^T layout
    #pragma unroll
    for (int s = 0; s < 2; ++s) {
      v16bf a = ldsA_frag(sP + (wave * 16 + lo) * 64 + s * 32, hi);
      #pragma unroll
      for (int ni = 0; ni < 4; ++ni) {
        v16bf bf = ldsB_frag(sVt + (ni * 16 + lo) * 64 + s * 32 + hi * 16);
        acc_o[ni] = wmma_bf16(a, bf, acc_o[ni]);
      }
    }
    __syncthreads();
  }

  // normalize + store O in [b][t][h*64+d]
  float invl[8];
  #pragma unroll
  for (int v = 0; v < 8; ++v) invl[v] = 1.f / l_st[v];
  #pragma unroll
  for (int ni = 0; ni < 4; ++ni)
    #pragma unroll
    for (int v = 0; v < 8; ++v) {
      int trow = brow + qt * 64 + wave * 16 + v + 8 * hi;
      O[(size_t)trow * (HH * 64) + h * 64 + ni * 16 + lo] = acc_o[ni][v] * invl[v];
    }
}

// ---------------------------------------------------------------------------
// Router: per-token softmax over 8 experts, top-2 renormalized combine weights
// ---------------------------------------------------------------------------
__global__ __launch_bounds__(256) void k_router(
    const float* __restrict__ xn, const float* __restrict__ rw,
    float* __restrict__ probs, float* __restrict__ comb) {
  int t = blockIdx.x * blockDim.x + threadIdx.x;
  if (t >= MROWS) return;
  float acc[EE];
  #pragma unroll
  for (int e = 0; e < EE; ++e) acc[e] = 0.f;
  const float* xr = xn + (size_t)t * CC;
  for (int k = 0; k < CC; ++k) {
    float xv = xr[k];
    const float* r = rw + k * EE;
    #pragma unroll
    for (int e = 0; e < EE; ++e) acc[e] += xv * r[e];
  }
  float m = acc[0];
  #pragma unroll
  for (int e = 1; e < EE; ++e) m = fmaxf(m, acc[e]);
  float sum = 0.f;
  #pragma unroll
  for (int e = 0; e < EE; ++e) { acc[e] = __expf(acc[e] - m); sum += acc[e]; }
  float inv = 1.f / sum;
  #pragma unroll
  for (int e = 0; e < EE; ++e) { acc[e] *= inv; probs[t * EE + e] = acc[e]; }
  int i1 = 0; float p1 = acc[0];
  #pragma unroll
  for (int e = 1; e < EE; ++e) if (acc[e] > p1) { p1 = acc[e]; i1 = e; }
  int i2 = -1; float p2 = -1.f;
  #pragma unroll
  for (int e = 0; e < EE; ++e) if (e != i1 && acc[e] > p2) { p2 = acc[e]; i2 = e; }
  float z = 1.f / (p1 + p2);
  #pragma unroll
  for (int e = 0; e < EE; ++e)
    comb[t * EE + e] = (e == i1) ? p1 * z : ((e == i2) ? p2 * z : 0.f);
}

// ---------------------------------------------------------------------------
// Aux loss: aux = E * sum_e f_e * mean_probs_e; deterministic reduction
// ---------------------------------------------------------------------------
__global__ __launch_bounds__(256) void k_aux(
    const float* __restrict__ probs, const float* __restrict__ comb,
    float* __restrict__ aux_out) {
  int tid = threadIdx.x;
  float fl[EE], pl[EE];
  #pragma unroll
  for (int e = 0; e < EE; ++e) { fl[e] = 0.f; pl[e] = 0.f; }
  for (int t = tid; t < MROWS; t += 256) {
    #pragma unroll
    for (int e = 0; e < EE; ++e) {
      fl[e] += (comb[t * EE + e] > 0.f) ? 1.f : 0.f;
      pl[e] += probs[t * EE + e];
    }
  }
  __shared__ float red[256];
  float Fs[EE], Ps[EE];
  for (int e = 0; e < EE; ++e) {
    red[tid] = fl[e]; __syncthreads();
    for (int s = 128; s > 0; s >>= 1) { if (tid < s) red[tid] += red[tid + s]; __syncthreads(); }
    Fs[e] = red[0]; __syncthreads();
    red[tid] = pl[e]; __syncthreads();
    for (int s = 128; s > 0; s >>= 1) { if (tid < s) red[tid] += red[tid + s]; __syncthreads(); }
    Ps[e] = red[0]; __syncthreads();
  }
  if (tid == 0) {
    float aux = 0.f;
    for (int e = 0; e < EE; ++e)
      aux += (Fs[e] / (float)(MROWS * 2)) * (Ps[e] / (float)MROWS);
    aux_out[0] = (float)EE * aux;
  }
}

// ---------------------------------------------------------------------------
// Launch
// ---------------------------------------------------------------------------
extern "C" void kernel_launch(void* const* d_in, const int* in_sizes, int n_in,
                              void* d_out, int out_size, void* d_ws, size_t ws_size,
                              hipStream_t stream) {
  const float* x        = (const float*)d_in[0];
  const float* cosT     = (const float*)d_in[1];
  const float* sinT     = (const float*)d_in[2];
  const float* ln1_w    = (const float*)d_in[3];
  const float* q_w      = (const float*)d_in[4];
  const float* k_w      = (const float*)d_in[5];
  const float* v_w      = (const float*)d_in[6];
  const float* o_w      = (const float*)d_in[7];
  const float* ln2_w    = (const float*)d_in[8];
  const float* router_w = (const float*)d_in[9];
  const float* W1       = (const float*)d_in[10];
  const float* W2       = (const float*)d_in[11];
  const float* W3       = (const float*)d_in[12];
  float* dout = (float*)d_out;

  // workspace layout (256B aligned)
  char* wsb = (char*)d_ws;
  size_t off = 0;
  auto alloc = [&](size_t bytes) { size_t o = off; off += (bytes + 255) & ~(size_t)255; return o; };
  float*  xn1   = (float*)(wsb + alloc((size_t)MROWS * CC * 4));
  float*  qf    = (float*)(wsb + alloc((size_t)MROWS * HH * 64 * 4));
  float*  kf    = (float*)(wsb + alloc((size_t)MROWS * KVH * 64 * 4));
  float*  vf    = (float*)(wsb + alloc((size_t)MROWS * KVH * 64 * 4));
  __bf16* qb    = (__bf16*)(wsb + alloc((size_t)MROWS * HH * 64 * 2));
  __bf16* kb    = (__bf16*)(wsb + alloc((size_t)MROWS * KVH * 64 * 2));
  float*  attn  = (float*)(wsb + alloc((size_t)MROWS * CC * 4));
  float*  x2    = (float*)(wsb + alloc((size_t)MROWS * CC * 4));
  float*  xn2   = (float*)(wsb + alloc((size_t)MROWS * CC * 4));
  float*  probs = (float*)(wsb + alloc((size_t)MROWS * EE * 4));
  float*  comb  = (float*)(wsb + alloc((size_t)MROWS * EE * 4));
  float*  g     = (float*)(wsb + alloc((size_t)MROWS * FF * 4));

  dim3 blk128(128);

  // 1) pre-attention RMSNorm
  k_rmsnorm<<<MROWS, 256, 0, stream>>>(x, ln1_w, xn1, nullptr, CC);

  // 2) QKV projections (bf16 WMMA GEMM)
  k_gemm<GM_STORE><<<dim3(HH * 64 / 64, MROWS / 64), blk128, 0, stream>>>(
      xn1, q_w, nullptr, qf, nullptr, MROWS, HH * 64, CC, 0, 0);
  k_gemm<GM_STORE><<<dim3(KVH * 64 / 64, MROWS / 64), blk128, 0, stream>>>(
      xn1, k_w, nullptr, kf, nullptr, MROWS, KVH * 64, CC, 0, 0);
  k_gemm<GM_STORE><<<dim3(KVH * 64 / 64, MROWS / 64), blk128, 0, stream>>>(
      xn1, v_w, nullptr, vf, nullptr, MROWS, KVH * 64, CC, 0, 0);

  // 3) RoPE -> bf16 Q/K
  {
    int nq = MROWS * HH * 32;
    k_rope<<<(nq + 255) / 256, 256, 0, stream>>>(qf, qb, cosT, sinT, MROWS, HH);
    int nk = MROWS * KVH * 32;
    k_rope<<<(nk + 255) / 256, 256, 0, stream>>>(kf, kb, cosT, sinT, MROWS, KVH);
  }

  // 4) windowed flash attention (WMMA + async LDS staging)
  k_attn<<<dim3(TT / 64, HH, BB), blk128, 0, stream>>>(qb, kb, vf, attn);

  // 5) output projection + residual -> x2
  k_gemm<GM_ADD><<<dim3(CC / 64, MROWS / 64), blk128, 0, stream>>>(
      attn, o_w, nullptr, x2, x, MROWS, CC, CC, 0, 0);

  // 6) post-attention RMSNorm; fuse copy of residual x2 into d_out
  k_rmsnorm<<<MROWS, 256, 0, stream>>>(x2, ln2_w, xn2, dout, CC);

  // 7) router
  k_router<<<MROWS / 256, 256, 0, stream>>>(xn2, router_w, probs, comb);

  // 8) MoE experts (dense, matching reference): g = silu(x@W2)*(x@W1);
  //    d_out += comb[:,e] * (g @ W3[e])
  for (int e = 0; e < EE; ++e) {
    const float* w1e = W1 + (size_t)e * CC * FF;
    const float* w2e = W2 + (size_t)e * CC * FF;
    const float* w3e = W3 + (size_t)e * FF * CC;
    k_gemm<GM_GATE><<<dim3(FF / 64, MROWS / 64), blk128, 0, stream>>>(
        xn2, w1e, w2e, g, nullptr, MROWS, FF, CC, 0, 0);
    k_gemm<GM_SCALE_ADD><<<dim3(CC / 64, MROWS / 64), blk128, 0, stream>>>(
        g, w3e, nullptr, dout, comb, MROWS, CC, FF, EE, e);
  }

  // 9) aux loss -> last output element
  k_aux<<<1, 256, 0, stream>>>(probs, comb, dout + (out_size - 1));
}
